// MoE_13795434955006
// MI455X (gfx1250) — matrix operand order repaired and checked by reading, not compiled, not run on hardware
//
#include <hip/hip_runtime.h>
#include <hip/hip_bf16.h>
#include <math.h>

// Problem constants (match reference)
#define NTOK 4096
#define DIN  1024
#define HID  2048
#define NEXP 16
#define TOPK 4
#define KC   32          // K-chunk per pipeline stage (one wmma K-depth)

// ---------- CDNA5 WMMA types ----------
typedef __attribute__((ext_vector_type(16))) __bf16 v16bf;
typedef __attribute__((ext_vector_type(8)))  float  v8f;

union Frag {            // 16 bf16 = 8 dwords = two b128 LDS loads
    uint4 u[2];
    v16bf v;
};

// f32 -> bf16 (RNE); let clang pick the best cvt sequence/instruction
__device__ __forceinline__ unsigned int pk2(float a, float b) {
    __bf16 ba = (__bf16)a;
    __bf16 bb = (__bf16)b;
    unsigned short ua, ub;
    __builtin_memcpy(&ua, &ba, 2);
    __builtin_memcpy(&ub, &bb, 2);
    return (unsigned int)ua | ((unsigned int)ub << 16);
}

// ---------- Kernel 1: zero the output (atomic-accumulated later) ----------
__global__ void zero_f4(float4* __restrict__ p, int n4) {
    int i = blockIdx.x * blockDim.x + threadIdx.x;
    if (i < n4) p[i] = make_float4(0.f, 0.f, 0.f, 0.f);
}

// ---------- Kernel 2: noisy top-k gating ----------
__global__ __launch_bounds__(64) void gate_kernel(
    const float* __restrict__ x, const float* __restrict__ neps,
    const float* __restrict__ wg, const float* __restrict__ wnz,
    float* __restrict__ gates)
{
    const int n = blockIdx.x;
    const int t = threadIdx.x;
    const int e = t & 15, seg = t >> 4;

    __shared__ float pc[64];
    __shared__ float pn[64];
    __shared__ float lg[16];

    const float* xr = x + (size_t)n * DIN;
    float ac = 0.f, an = 0.f;
    #pragma unroll 4
    for (int i = 0; i < DIN / 4; i++) {
        int d = seg * (DIN / 4) + i;
        float xv = xr[d];
        ac += xv * wg [d * NEXP + e];
        an += xv * wnz[d * NEXP + e];
    }
    pc[t] = ac; pn[t] = an;
    __syncthreads();

    if (t < NEXP) {
        float c = pc[t] + pc[16 + t] + pc[32 + t] + pc[48 + t];
        float v = pn[t] + pn[16 + t] + pn[32 + t] + pn[48 + t];
        float sp = (v > 15.f) ? v : log1pf(expf(v));   // stable softplus
        float sd = sp + 1e-2f;
        lg[t] = c + neps[n * NEXP + t] * sd;
    }
    __syncthreads();

    if (t == 0) {
        float lv[NEXP];
        #pragma unroll
        for (int j = 0; j < NEXP; j++) lv[j] = lg[j];
        int   ti[TOPK]; float tv[TOPK];
        #pragma unroll
        for (int k = 0; k < TOPK; k++) {
            int bi = 0; float bv = -3.4e38f;
            #pragma unroll
            for (int j = 0; j < NEXP; j++)
                if (lv[j] > bv) { bv = lv[j]; bi = j; }
            ti[k] = bi; tv[k] = bv; lv[bi] = -3.4e38f;
        }
        float m = tv[0];
        float ex[TOPK], s = 0.f;
        #pragma unroll
        for (int k = 0; k < TOPK; k++) { ex[k] = expf(tv[k] - m); s += ex[k]; }
        float out[NEXP];
        #pragma unroll
        for (int j = 0; j < NEXP; j++) out[j] = 0.f;
        #pragma unroll
        for (int k = 0; k < TOPK; k++) out[ti[k]] = ex[k] / s;
        #pragma unroll
        for (int j = 0; j < NEXP; j++) gates[n * NEXP + j] = out[j];
    }
}

// ---------- Kernel 3: deterministic per-expert token compaction ----------
__global__ __launch_bounds__(32) void build_lists(
    const float* __restrict__ gates,
    int* __restrict__ tok_list, float* __restrict__ gate_list,
    int* __restrict__ cnt)
{
    const int e = blockIdx.x;
    const int lane = threadIdx.x;
    int base = 0;
    for (int c0 = 0; c0 < NTOK; c0 += 32) {
        int tok = c0 + lane;
        float g = gates[tok * NEXP + e];
        bool act = (g > 0.f);
        unsigned long long mask = __ballot(act);        // wave32: upper bits 0
        int pre = __popcll(mask & ((1ull << lane) - 1ull));
        if (act) {
            tok_list [e * NTOK + base + pre] = tok;
            gate_list[e * NTOK + base + pre] = g;
        }
        base += __popcll(mask);
    }
    if (lane == 0) cnt[e] = base;
}

// ---------- Kernel 4: routed grouped GEMM via bf16 WMMA ----------
// 128 threads = 4 waves (wave32), block tile 128 tokens x 128 H cols,
// each wave owns a 64x64 sub-tile = 4x4 wmma_f32_16x16x32_bf16 accumulators.
// Double-buffered LDS + register prefetch: one barrier per K-chunk; next
// chunk's global loads are issued before the WMMA block so memory latency
// overlaps matrix math.
__global__ __launch_bounds__(128) void moe_gemm(
    const float* __restrict__ x, const float* __restrict__ ew,
    const float* __restrict__ eb,
    const int* __restrict__ tok_list, const float* __restrict__ gate_list,
    const int* __restrict__ cnt, float* __restrict__ y)
{
    const int e     = blockIdx.z;
    const int mbase = blockIdx.y * 128;
    const int h0    = blockIdx.x * 128;
    const int count = cnt[e];
    if (mbase >= count) return;

    // Row stride 40 ushorts = 80 B (16B aligned for b128 fragment loads)
    __shared__ __align__(16) unsigned short lA[2][128][40];  // [token][k] bf16
    __shared__ __align__(16) unsigned short lB[2][128][40];  // [h][k] bf16 (transposed)
    __shared__ int   sIdx [128];
    __shared__ float sGate[128];
    __shared__ float sBias[128];

    const int t = threadIdx.x;
    {
        int r = mbase + t;
        int gi = 0; float gg = 0.f;
        if (r < count) { gi = tok_list[e * NTOK + r]; gg = gate_list[e * NTOK + r]; }
        sIdx[t] = gi; sGate[t] = gg;                     // padded rows: gate 0
        sBias[t] = eb[e * HID + h0 + t];
    }
    __syncthreads();

    const int lane  = t & 31;
    const int w     = t >> 5;          // 4 waves, 2x2 arrangement
    const int wmoff = (w >> 1) * 64;   // 0 / 64
    const int wnoff = (w & 1) * 64;    // 0 / 64
    const int khalf = lane >> 4;       // ISA 16-bit fragment half
    const int l16   = lane & 15;

    v8f acc[4][4] = {};
    const float* We = ew + (size_t)e * DIN * HID;

    // Staging assignments (128 threads, 32 elements each per matrix)
    const int ar = t;                        // A: 1 thread/row, all 32 k
    const int kp = t >> 3, hs = (t & 7) * 16;// B: k-row pair 2kp, 16 h each
    const float* xrow = x + (size_t)sIdx[ar] * DIN;

    float4 pa[8];          // prefetched A (32 f32)
    float4 pb0[4], pb1[4]; // prefetched B rows 2kp / 2kp+1 (16 f32 each)

    // ---- load + store chunk 0 ----
    {
        const float4* sA = (const float4*)(xrow);
        #pragma unroll
        for (int i = 0; i < 8; i++) pa[i] = sA[i];
        const float4* s0 = (const float4*)(We + (size_t)(2 * kp    ) * HID + h0 + hs);
        const float4* s1 = (const float4*)(We + (size_t)(2 * kp + 1) * HID + h0 + hs);
        #pragma unroll
        for (int i = 0; i < 4; i++) { pb0[i] = s0[i]; pb1[i] = s1[i]; }
    }
    {
        unsigned int* dst = (unsigned int*)&lA[0][ar][0];
        #pragma unroll
        for (int i = 0; i < 8; i++) {
            dst[2*i]   = pk2(pa[i].x, pa[i].y);
            dst[2*i+1] = pk2(pa[i].z, pa[i].w);
        }
        const float* f0 = (const float*)pb0;
        const float* f1 = (const float*)pb1;
        #pragma unroll
        for (int i = 0; i < 16; i++)
            ((unsigned int*)&lB[0][hs + i][0])[kp] = pk2(f0[i], f1[i]);
    }
    __syncthreads();

    int cur = 0;
    for (int ck = 0; ck < DIN / KC; ck++) {
        const bool more = (ck + 1) < (DIN / KC);

        // ---- issue next chunk's global loads (overlaps with WMMA below) ----
        if (more) {
            int kn = (ck + 1) * KC;
            const float4* sA = (const float4*)(xrow + kn);
            #pragma unroll
            for (int i = 0; i < 8; i++) pa[i] = sA[i];
            const float4* s0 = (const float4*)(We + (size_t)(kn + 2 * kp    ) * HID + h0 + hs);
            const float4* s1 = (const float4*)(We + (size_t)(kn + 2 * kp + 1) * HID + h0 + hs);
            #pragma unroll
            for (int i = 0; i < 4; i++) { pb0[i] = s0[i]; pb1[i] = s1[i]; }
        }

        // ---- fragment loads (ISA 7.12.2 16-bit layouts) + 16 WMMAs ----
        Frag a[4], b[4];
        #pragma unroll
        for (int mt = 0; mt < 4; mt++) {
            int row = wmoff + mt * 16 + l16;
            a[mt].u[0] = *(const uint4*)&lA[cur][row][ 8 * khalf];       // K = 8kh..+7
            a[mt].u[1] = *(const uint4*)&lA[cur][row][16 + 8 * khalf];   // K = 16+8kh..+7
        }
        #pragma unroll
        for (int nt = 0; nt < 4; nt++) {
            int col = wnoff + nt * 16 + l16;
            b[nt].u[0] = *(const uint4*)&lB[cur][col][ 8 * khalf];       // K = 8kh..+7
            b[nt].u[1] = *(const uint4*)&lB[cur][col][16 + 8 * khalf];   // K = 16+8kh..+7
        }
        #pragma unroll
        for (int mt = 0; mt < 4; mt++)
            #pragma unroll
            for (int nt = 0; nt < 4; nt++)
                acc[mt][nt] = __builtin_amdgcn_wmma_f32_16x16x32_bf16(
                    false, a[mt].v, false, b[nt].v,
                    (short)0, acc[mt][nt], false, false);

        // ---- stage next chunk into the other buffer, single barrier ----
        if (more) {
            int nxt = cur ^ 1;
            unsigned int* dst = (unsigned int*)&lA[nxt][ar][0];
            #pragma unroll
            for (int i = 0; i < 8; i++) {
                dst[2*i]   = pk2(pa[i].x, pa[i].y);
                dst[2*i+1] = pk2(pa[i].z, pa[i].w);
            }
            const float* f0 = (const float*)pb0;
            const float* f1 = (const float*)pb1;
            #pragma unroll
            for (int i = 0; i < 16; i++)
                ((unsigned int*)&lB[nxt][hs + i][0])[kp] = pk2(f0[i], f1[i]);
            __syncthreads();      // protects buf 'cur' overwrite next iteration
            cur = nxt;
        }
    }

    // ---- epilogue: y[tok][h] += gate * (acc + bias) ----
    #pragma unroll
    for (int mt = 0; mt < 4; mt++) {
        #pragma unroll
        for (int nt = 0; nt < 4; nt++) {
            int ncol = wnoff + nt * 16 + l16;
            int hcol = h0 + ncol;
            float bias = sBias[ncol];
            #pragma unroll
            for (int r = 0; r < 8; r++) {
                int row = wmoff + mt * 16 + khalf * 8 + r;   // C/D layout: M = r + 8*kh
                float g = sGate[row];
                if (g != 0.f) {
                    float val = g * (acc[mt][nt][r] + bias);
                    atomicAdd(&y[(size_t)sIdx[row] * HID + hcol], val);
                }
            }
        }
    }
}

// ---------- host launcher ----------
extern "C" void kernel_launch(void* const* d_in, const int* in_sizes, int n_in,
                              void* d_out, int out_size, void* d_ws, size_t ws_size,
                              hipStream_t stream) {
    const float* x    = (const float*)d_in[0];
    const float* neps = (const float*)d_in[1];
    const float* wg   = (const float*)d_in[2];
    const float* wnz  = (const float*)d_in[3];
    const float* ew   = (const float*)d_in[4];
    const float* eb   = (const float*)d_in[5];
    float* y = (float*)d_out;

    // workspace layout (~770 KB)
    char* ws = (char*)d_ws;
    float* gates     = (float*)ws;                               // N*E f32
    int*   cnt       = (int*)  (ws + 256 * 1024);                // E ints
    int*   tok_list  = (int*)  (ws + 256 * 1024 + 1024);         // E*N ints
    float* gate_list = (float*)(ws + 256 * 1024 + 1024 + 256 * 1024); // E*N f32

    const int n4 = (NTOK * HID) / 4;
    zero_f4<<<(n4 + 255) / 256, 256, 0, stream>>>((float4*)y, n4);
    gate_kernel<<<NTOK, 64, 0, stream>>>(x, neps, wg, wnz, gates);
    build_lists<<<NEXP, 32, 0, stream>>>(gates, tok_list, gate_list, cnt);
    dim3 grid(HID / 128, NTOK / 128, NEXP);
    moe_gemm<<<grid, 128, 0, stream>>>(x, ew, eb, tok_list, gate_list, cnt, y);
}
